// FeatureFusionModel_53867479826851
// MI455X (gfx1250) — matrix operand order repaired and checked by reading, not compiled, not run on hardware
//
#include <hip/hip_runtime.h>
#include <hip/hip_bf16.h>
#include <stdint.h>

// ---------------- problem constants (from reference) ----------------
#define B_     2
#define V_     65536
#define NV_    6
#define DIM_   384      // token dim
#define PF_    64       // voxel feature dim
#define FUSE_  448      // PF_ + DIM_
#define HID_   256
#define OUT_   16
#define MTOK_  1369     // 37*37
#define GRID_  37

#define MTILE_ 64       // voxel rows per block (4 x 16-row WMMA tiles)

// packed-weight buffer element counts (u32 each holding 2 bf16)
#define W1P_N  (14 * 16 * 32 * 8)   // kt(14) x nt(16) x lane(32) x vgpr(8)
#define W2P_N  (8 * 32 * 8)         // kt(8)  x lane(32) x vgpr(8)

// LDS row strides (odd dword counts -> conflict-free across 64 banks)
#define ASTR_U16 450                // 225 dwords per fused row
#define HSTR_U16 258                // 129 dwords per hidden row

typedef __attribute__((ext_vector_type(16))) __bf16 v16bf;
typedef __attribute__((ext_vector_type(8)))  float  v8f;

union Frag {
  unsigned int u[8];
  uint4        q[2];
  v16bf        v;
};

static __device__ __forceinline__ unsigned short f2bf(float f) {
  // round-to-nearest-even fp32 -> bf16
  unsigned int u = __float_as_uint(f);
  u += 0x7FFFu + ((u >> 16) & 1u);
  return (unsigned short)(u >> 16);
}

// ---------------- kernel 1: mean over views -> packed bf16 ----------------
__global__ void mean_tokens_bf16_kernel(const float* __restrict__ pt,
                                        unsigned int* __restrict__ outp) {
  int j = blockIdx.x * blockDim.x + threadIdx.x;
  const int total = B_ * MTOK_ * (DIM_ / 2);
  if (j >= total) return;
  int dp = j % (DIM_ / 2);
  int m  = (j / (DIM_ / 2)) % MTOK_;
  int b  = j / ((DIM_ / 2) * MTOK_);
  int d  = dp * 2;
  float s0 = 0.f, s1 = 0.f;
#pragma unroll
  for (int nv = 0; nv < NV_; ++nv) {
    const float* p = pt + (((size_t)(b * NV_ + nv) * MTOK_ + m) * DIM_ + d);
    s0 += p[0];
    s1 += p[1];
  }
  s0 *= (1.f / 6.f);
  s1 *= (1.f / 6.f);
  outp[j] = (unsigned int)f2bf(s0) | ((unsigned int)f2bf(s1) << 16);  // [B][M][DIM/2]
}

// ---------------- kernel 2: pre-swizzle W1 / W2 into B-fragment layout ----------------
// B-fragment, this code's k order (same as A, not AMD's table): lane l (n = l&15), half = l>>4,
// VGPR i holds K = half*16 + 2*i and K+1. Stored lane-major: [tile][lane][i] u32.
__global__ void pack_weights_kernel(const float* __restrict__ W1,
                                    const float* __restrict__ W2,
                                    unsigned int* __restrict__ w1p,
                                    unsigned int* __restrict__ w2p) {
  int j = blockIdx.x * blockDim.x + threadIdx.x;
  if (j < W1P_N) {
    int i  = j & 7;
    int l  = (j >> 3) & 31;
    int t  = j >> 8;        // kt*16 + nt
    int nt = t & 15;
    int kt = t >> 4;
    int half = l >> 4;
    int n  = nt * 16 + (l & 15);
    int k  = kt * 32 + half * 16 + 2 * i;
    float lo = W1[(size_t)k * HID_ + n];
    float hi = W1[(size_t)(k + 1) * HID_ + n];
    w1p[j] = (unsigned int)f2bf(lo) | ((unsigned int)f2bf(hi) << 16);
  } else if (j < W1P_N + W2P_N) {
    int jj = j - W1P_N;
    int i  = jj & 7;
    int l  = (jj >> 3) & 31;
    int kt = jj >> 8;
    int half = l >> 4;
    int n  = l & 15;
    int k  = kt * 32 + half * 16 + 2 * i;
    float lo = W2[(size_t)k * OUT_ + n];
    float hi = W2[(size_t)(k + 1) * OUT_ + n];
    w2p[jj] = (unsigned int)f2bf(lo) | ((unsigned int)f2bf(hi) << 16);
  }
}

// ---------------- kernel 3: project + gather + MLP via WMMA ----------------
// One block = 64 voxels (4 x 16-row tiles). 8 waves: wave w owns N-tiles 2w, 2w+1.
// Each B fragment loaded once per k-step is reused by 4 M-tiles (4x L2 traffic cut).
__global__ __launch_bounds__(256) void fusion_mlp_wmma_kernel(
    const float* __restrict__ vfeat, const float* __restrict__ vcoord,
    const float* __restrict__ Km,    const float* __restrict__ Rt,
    const unsigned int* __restrict__ meanp,
    const unsigned int* __restrict__ w1p,
    const unsigned int* __restrict__ w2p,
    const float* __restrict__ b1,    const float* __restrict__ b2,
    float* __restrict__ outp) {
  __shared__ __align__(16) unsigned short ldsA[MTILE_ * ASTR_U16];  // 57,600 B
  __shared__ __align__(16) unsigned short ldsH[MTILE_ * HSTR_U16];  // 33,024 B
  __shared__ int ldsIdx[MTILE_];

  const int tid = threadIdx.x;
  const int blk = blockIdx.x;
  const int b   = blk / (V_ / MTILE_);
  const int v0  = (blk % (V_ / MTILE_)) * MTILE_;

  // ---- phase 1: projection -> patch index for the 64 rows ----
  if (tid < MTILE_) {
    const float* c = vcoord + (size_t)(b * V_ + v0 + tid) * 3;
    float x = c[0], y = c[1], z = c[2];
    float c0 = Rt[0] * x + Rt[1] * y + Rt[2]  * z + Rt[3];
    float c1 = Rt[4] * x + Rt[5] * y + Rt[6]  * z + Rt[7];
    float c2 = Rt[8] * x + Rt[9] * y + Rt[10] * z + Rt[11];
    float p0 = Km[0] * c0 + Km[1] * c1 + Km[2] * c2;
    float p1 = Km[3] * c0 + Km[4] * c1 + Km[5] * c2;
    float p2 = Km[6] * c0 + Km[7] * c1 + Km[8] * c2;
    float inv = 1.f / (p2 + 1e-6f);
    float uu = p0 * inv * (518.f / 600.f);
    float vv = p1 * inv * (518.f / 900.f);
    int px = (int)(uu * (1.f / 14.f));   // trunc toward zero, like .long()
    int py = (int)(vv * (1.f / 14.f));
    px = px < 0 ? 0 : (px > GRID_ - 1 ? GRID_ - 1 : px);
    py = py < 0 ? 0 : (py > GRID_ - 1 ? GRID_ - 1 : py);
    ldsIdx[tid] = px * GRID_ + py;
  }
  __syncthreads();

  // ---- phase 2: stage fused [64 x 448] bf16 rows into LDS ----
  unsigned int* ldsA32 = (unsigned int*)ldsA;
  // voxel features: 64 rows x 64 floats, float4 loads
  for (int j = tid; j < MTILE_ * (PF_ / 4); j += 256) {
    int row = j >> 4;           // PF_/4 == 16 per row
    int c4  = (j & 15) * 4;
    float4 f = *(const float4*)(vfeat + (size_t)(b * V_ + v0 + row) * PF_ + c4);
    unsigned short* dst = ldsA + row * ASTR_U16 + c4;
    dst[0] = f2bf(f.x); dst[1] = f2bf(f.y); dst[2] = f2bf(f.z); dst[3] = f2bf(f.w);
  }
  // gathered tokens: 64 rows x 192 u32, uint4 loads (bf16 already packed)
  for (int j = tid; j < MTILE_ * (DIM_ / 8); j += 256) {
    int row = j / (DIM_ / 8);   // 48 uint4 per row
    int c4  = (j - row * (DIM_ / 8)) * 4;
    uint4 t = *(const uint4*)(meanp + ((size_t)b * MTOK_ + ldsIdx[row]) * (DIM_ / 2) + c4);
    unsigned int* dst = ldsA32 + row * (ASTR_U16 / 2) + (PF_ / 2) + c4;
    dst[0] = t.x; dst[1] = t.y; dst[2] = t.z; dst[3] = t.w;
  }
  __syncthreads();

  // ---- phase 3: GEMM1  h = relu(A[64x448] @ W1[448x256] + b1) ----
  const int wave = tid >> 5;
  const int lane = tid & 31;
  const int half = lane >> 4;
  const int am   = lane & 15;     // A row within tile / output column
  const int nt0  = wave * 2;
  const int nt1  = wave * 2 + 1;

  v8f acc[4][2];
#pragma unroll
  for (int mt = 0; mt < 4; ++mt)
#pragma unroll
    for (int c = 0; c < 2; ++c)
      acc[mt][c] = (v8f){0.f, 0.f, 0.f, 0.f, 0.f, 0.f, 0.f, 0.f};

#pragma unroll 1
  for (int kt = 0; kt < FUSE_ / 32; ++kt) {
    Frag bf0, bf1;
    const uint4* p0 = (const uint4*)(w1p + ((size_t)(kt * 16 + nt0) * 32 + lane) * 8);
    bf0.q[0] = p0[0]; bf0.q[1] = p0[1];
    const uint4* p1 = (const uint4*)(w1p + ((size_t)(kt * 16 + nt1) * 32 + lane) * 8);
    bf1.q[0] = p1[0]; bf1.q[1] = p1[1];
#pragma unroll
    for (int mt = 0; mt < 4; ++mt) {
      Frag a;
#pragma unroll
      for (int i = 0; i < 8; ++i) {
        int k = kt * 32 + half * 16 + 2 * i;
        a.u[i] = ldsA32[(mt * 16 + am) * (ASTR_U16 / 2) + (k >> 1)];
      }
      acc[mt][0] = __builtin_amdgcn_wmma_f32_16x16x32_bf16(
          false, a.v, false, bf0.v, (short)0, acc[mt][0], false, false);
      acc[mt][1] = __builtin_amdgcn_wmma_f32_16x16x32_bf16(
          false, a.v, false, bf1.v, (short)0, acc[mt][1], false, false);
    }
  }

  // epilogue: +bias, ReLU, bf16 -> LDS H
  float bias0 = b1[nt0 * 16 + am];
  float bias1 = b1[nt1 * 16 + am];
#pragma unroll
  for (int mt = 0; mt < 4; ++mt) {
#pragma unroll
    for (int r = 0; r < 8; ++r) {
      int m = mt * 16 + r + half * 8;
      float h0 = acc[mt][0][r] + bias0; h0 = h0 > 0.f ? h0 : 0.f;
      float h1 = acc[mt][1][r] + bias1; h1 = h1 > 0.f ? h1 : 0.f;
      ldsH[m * HSTR_U16 + nt0 * 16 + am] = f2bf(h0);
      ldsH[m * HSTR_U16 + nt1 * 16 + am] = f2bf(h1);
    }
  }
  __syncthreads();

  // ---- phase 4: GEMM2  out = H[64x256] @ W2[256x16] + b2  (waves 0-3) ----
  if (wave < 4) {
    unsigned int* ldsH32 = (unsigned int*)ldsH;
    const int mt = wave;            // 16-row tile of H
    v8f acc2 = {0.f, 0.f, 0.f, 0.f, 0.f, 0.f, 0.f, 0.f};
#pragma unroll
    for (int kt = 0; kt < HID_ / 32; ++kt) {
      Frag a;
#pragma unroll
      for (int i = 0; i < 8; ++i) {
        int k = kt * 32 + half * 16 + 2 * i;
        a.u[i] = ldsH32[(mt * 16 + am) * (HSTR_U16 / 2) + (k >> 1)];
      }
      Frag bb;
      const uint4* p = (const uint4*)(w2p + ((size_t)kt * 32 + lane) * 8);
      bb.q[0] = p[0]; bb.q[1] = p[1];
      acc2 = __builtin_amdgcn_wmma_f32_16x16x32_bf16(
          false, a.v, false, bb.v, (short)0, acc2, false, false);
    }
    float bias = b2[am];
    float* op = outp + (size_t)(b * V_ + v0 + mt * 16) * OUT_;
#pragma unroll
    for (int r = 0; r < 8; ++r) {
      int m = r + half * 8;
      op[m * OUT_ + am] = acc2[r] + bias;
    }
  }
}

// ---------------- launcher ----------------
extern "C" void kernel_launch(void* const* d_in, const int* in_sizes, int n_in,
                              void* d_out, int out_size, void* d_ws, size_t ws_size,
                              hipStream_t stream) {
  (void)in_sizes; (void)n_in; (void)out_size; (void)ws_size;
  const float* pt    = (const float*)d_in[0];  // patch_tokens  [2,6,1369,384]
  const float* vfeat = (const float*)d_in[1];  // voxel_features[2,65536,64]
  const float* vcoor = (const float*)d_in[2];  // voxel_coords  [2,65536,3]
  const float* Km    = (const float*)d_in[3];  // K  [3,3]
  const float* Rt    = (const float*)d_in[4];  // Rt [3,4]
  const float* W1    = (const float*)d_in[5];  // [448,256]
  const float* b1    = (const float*)d_in[6];  // [256]
  const float* W2    = (const float*)d_in[7];  // [256,16]
  const float* b2    = (const float*)d_in[8];  // [16]
  float* outp = (float*)d_out;                 // [2,65536,16]

  // workspace carve-out (all 256B-aligned)
  const size_t MEAN_BYTES = (size_t)B_ * MTOK_ * DIM_ * 2;  // 2,102,784 B
  unsigned int* meanp = (unsigned int*)d_ws;
  unsigned int* w1p   = (unsigned int*)((char*)d_ws + MEAN_BYTES);
  unsigned int* w2p   = w1p + W1P_N;

  {
    int total = B_ * MTOK_ * (DIM_ / 2);
    mean_tokens_bf16_kernel<<<(total + 255) / 256, 256, 0, stream>>>(pt, meanp);
  }
  {
    int total = W1P_N + W2P_N;
    pack_weights_kernel<<<(total + 255) / 256, 256, 0, stream>>>(W1, W2, w1p, w2p);
  }
  {
    int blocks = B_ * (V_ / MTILE_);  // 2048
    fusion_mlp_wmma_kernel<<<blocks, 256, 0, stream>>>(vfeat, vcoor, Km, Rt,
                                                       meanp, w1p, w2p, b1, b2, outp);
  }
}
